// MultiHeadAttention_18270790877726
// MI455X (gfx1250) — matrix operand order repaired and checked by reading, not compiled, not run on hardware
//
#include <hip/hip_runtime.h>
#include <hip/hip_bf16.h>

#define NH 16

typedef __attribute__((ext_vector_type(16))) __bf16 v16bf;
typedef __attribute__((ext_vector_type(8)))  float  v8f;

union ABfrag { v16bf v; uint4 u[2]; __bf16 h[16]; };
union CDfrag { v8f v; float f[8]; };

__device__ __forceinline__ v8f wmma_bf16(v16bf a, v16bf b, v8f c) {
  // 8 args: (neg_a, A, neg_b, B, c_mod, C, reuse_a, reuse_b)
  return __builtin_amdgcn_wmma_f32_16x16x32_bf16(false, a, false, b, (short)0, c,
                                                 false, false);
}

// ---------------------------------------------------------------- cast kernel
__global__ void cast_f32_bf16(const float* __restrict__ in, __bf16* __restrict__ out,
                              size_t n) {
  size_t i = (size_t)blockIdx.x * blockDim.x + threadIdx.x;
  size_t stride = (size_t)gridDim.x * blockDim.x;
  for (; i < n; i += stride) out[i] = (__bf16)in[i];
}

// ------------------------------------------------------- generic C = A * W^T
// A: [M,K] bf16 row-major.  W: [N,K] bf16 row-major (torch Linear weight).
// mode 0: C bf16 row-major [M,N]
// mode 1: C bf16 transposed per batch: [B][N][klen], b = m / klen (for V)
// mode 2: C f32 row-major [M,N]
// Each wave computes a 64x64 tile of C with 16 accumulators.
__global__ __launch_bounds__(256)
void gemm_bf16_wmma(const __bf16* __restrict__ A, const __bf16* __restrict__ W,
                    void* __restrict__ Cout, int M, int N, int K,
                    int mode, int klen) {
  const int lane = threadIdx.x & 31;
  const int sub  = lane >> 4;   // half-wave: 0 or 1
  const int l16  = lane & 15;
  const int wavesPerBlk = blockDim.x >> 5;
  const int wid = blockIdx.x * wavesPerBlk + (threadIdx.x >> 5);
  const int ntiles = N >> 6;
  const int mtiles = M >> 6;
  if (wid >= mtiles * ntiles) return;   // wave-uniform: EXEC stays all-ones
  const int tm = (wid / ntiles) << 6;
  const int tn = (wid % ntiles) << 6;

  v8f acc[4][4] = {};

  for (int k0 = 0; k0 < K; k0 += 32) {
    ABfrag a[4], b[4];
#pragma unroll
    for (int i = 0; i < 4; ++i) {
      // A fragment: row = tm+i*16+l16; elem j -> K = k0 + (j/8)*16 + sub*8 + j%8
      const __bf16* ap = A + (size_t)(tm + i * 16 + l16) * K + k0 + sub * 8;
      a[i].u[0] = *(const uint4*)ap;
      a[i].u[1] = *(const uint4*)(ap + 16);
    }
#pragma unroll
    for (int j = 0; j < 4; ++j) {
      // B fragment: col n = tn+j*16+l16; elem i -> K = k0 + (i/8)*16 + sub*8 + i%8
      const __bf16* wp = W + (size_t)(tn + j * 16 + l16) * K + k0 + sub * 8;
      b[j].u[0] = *(const uint4*)wp;
      b[j].u[1] = *(const uint4*)(wp + 16);
    }
#pragma unroll
    for (int i = 0; i < 4; ++i)
#pragma unroll
      for (int j = 0; j < 4; ++j)
        acc[i][j] = wmma_bf16(a[i].v, b[j].v, acc[i][j]);
  }

#pragma unroll
  for (int i = 0; i < 4; ++i) {
#pragma unroll
    for (int j = 0; j < 4; ++j) {
      CDfrag c; c.v = acc[i][j];
      const int col = tn + j * 16 + l16;
      const int row0 = tm + i * 16 + sub * 8;  // VGPR r -> row0 + r
      if (mode == 0) {
        __bf16* o = (__bf16*)Cout;
#pragma unroll
        for (int r = 0; r < 8; ++r)
          o[(size_t)(row0 + r) * N + col] = (__bf16)c.f[r];
      } else if (mode == 1) {
        const int bb = row0 / klen;            // tile never straddles batch
        const int ml = row0 - bb * klen;
        __bf16* o = (__bf16*)Cout + (size_t)bb * N * klen + (size_t)col * klen + ml;
        union { __bf16 h[8]; uint4 u; } t;
#pragma unroll
        for (int r = 0; r < 8; ++r) t.h[r] = (__bf16)c.f[r];
        *(uint4*)o = t.u;                      // 8 consecutive klen positions
      } else {
        float* o = (float*)Cout;
#pragma unroll
        for (int r = 0; r < 8; ++r)
          o[(size_t)(row0 + r) * N + col] = c.f[r];
      }
    }
  }
}

// --------------------------------------------------- flash-style relative MHA
// One wave handles (b, h, 16-row query tile). Q/K/R bf16 [rows, H]; V stored
// transposed [B][H][KLEN] so PV B-fragments are contiguous along keys.
__global__ __launch_bounds__(32)
void relattn_wmma(const __bf16* __restrict__ Q, const __bf16* __restrict__ Kp,
                  const __bf16* __restrict__ Vt, const __bf16* __restrict__ R,
                  const float* __restrict__ ubias, const float* __restrict__ vbias,
                  __bf16* __restrict__ Oout, int M, int KLEN, int H) {
  const int lane = threadIdx.x & 31;
  const int sub  = lane >> 4;
  const int l16  = lane & 15;
  const int mtiles = M >> 4;
  int blk = blockIdx.x;
  const int b = blk / (NH * mtiles); blk -= b * NH * mtiles;
  const int h = blk / mtiles;
  const int mt = (blk % mtiles) << 4;       // query-tile base (within M)
  const int hd0 = h * 64;
  const int memlen = KLEN - M;
  const float scale = 0.125f;               // 1/sqrt(64)

  __shared__ __align__(16) __bf16 Plds[16][48];  // 16x32 P tile (stride 48 keeps 16B align)

  // Content / positional query A-fragments (hd = 64 -> 2 K-chunks of 32)
  ABfrag qc[2], qp[2];
  {
    const __bf16* qptr = Q + (size_t)(b * M + mt + l16) * H + hd0;
    const float* up = ubias + h * 64;
    const float* vp = vbias + h * 64;
#pragma unroll
    for (int c = 0; c < 2; ++c)
#pragma unroll
      for (int i = 0; i < 16; ++i) {
        const int d = c * 32 + sub * 16 + i;
        const float q = (float)qptr[d];
        qc[c].h[i] = (__bf16)(q + up[d]);
        qp[c].h[i] = (__bf16)(q + vp[d]);
      }
  }

  v8f oacc[4] = {};
  float mrun[8], lrun[8];
#pragma unroll
  for (int r = 0; r < 8; ++r) { mrun[r] = -INFINITY; lrun[r] = 0.0f; }

  const int kmax = mt + 15 + memlen;        // last key any row in tile can see

  for (int kk0 = 0; kk0 < KLEN && kk0 <= kmax; kk0 += 32) {
    CDfrag s[2];
#pragma unroll
    for (int t = 0; t < 2; ++t) {
      const int kk = kk0 + t * 16;
      v8f cacc = {};
      if (kk <= kmax) {
        ABfrag kb0, kb1, rb0, rb1;
        const __bf16* kptr = Kp + (size_t)(b * KLEN + kk + l16) * H + hd0 + sub * 16;
        kb0.u[0] = *(const uint4*)kptr;        kb0.u[1] = *(const uint4*)(kptr + 8);
        kb1.u[0] = *(const uint4*)(kptr + 32); kb1.u[1] = *(const uint4*)(kptr + 40);
        const __bf16* rptr = R + (size_t)(kk + l16) * H + hd0 + sub * 16;
        rb0.u[0] = *(const uint4*)rptr;        rb0.u[1] = *(const uint4*)(rptr + 8);
        rb1.u[0] = *(const uint4*)(rptr + 32); rb1.u[1] = *(const uint4*)(rptr + 40);
        cacc = wmma_bf16(qp[1].v, rb1.v, cacc);
        cacc = wmma_bf16(qp[0].v, rb0.v, cacc);
        cacc = wmma_bf16(qc[1].v, kb1.v, cacc);
        cacc = wmma_bf16(qc[0].v, kb0.v, cacc);
      }
      s[t].v = cacc;
      const int kpos = kk + l16;
#pragma unroll
      for (int r = 0; r < 8; ++r) {
        const int mq = mt + r + sub * 8;
        s[t].f[r] = (kk <= kmax && kpos <= mq + memlen) ? s[t].f[r] * scale
                                                        : -INFINITY;
      }
    }

    // online softmax: a row spans the 16 lanes of one half-wave
    float p0[8], p1[8];
#pragma unroll
    for (int r = 0; r < 8; ++r) {
      float mx = fmaxf(s[0].f[r], s[1].f[r]);
#pragma unroll
      for (int off = 8; off; off >>= 1) mx = fmaxf(mx, __shfl_xor(mx, off, 16));
      const float mnew = fmaxf(mrun[r], mx);
      const float corr = __expf(mrun[r] - mnew);
      mrun[r] = mnew;
      p0[r] = __expf(s[0].f[r] - mnew);
      p1[r] = __expf(s[1].f[r] - mnew);
      float ps = p0[r] + p1[r];
#pragma unroll
      for (int off = 8; off; off >>= 1) ps += __shfl_xor(ps, off, 16);
      lrun[r] = lrun[r] * corr + ps;
#pragma unroll
      for (int t = 0; t < 4; ++t) oacc[t][r] *= corr;
    }

    // restage P (C-layout) -> A-fragment layout through LDS
#pragma unroll
    for (int r = 0; r < 8; ++r) {
      Plds[r + sub * 8][l16]      = (__bf16)p0[r];
      Plds[r + sub * 8][16 + l16] = (__bf16)p1[r];
    }
    __syncthreads();
    ABfrag pf;
    pf.u[0] = *(const uint4*)&Plds[l16][sub * 8];
    pf.u[1] = *(const uint4*)&Plds[l16][16 + sub * 8];

    // O(16x64) += P(16x32) @ V(32x64); Vt contiguous along key dimension
    const size_t vbase = ((size_t)b * H + hd0) * KLEN;
#pragma unroll
    for (int t = 0; t < 4; ++t) {
      ABfrag vf;
      const __bf16* vptr = Vt + vbase + (size_t)(t * 16 + l16) * KLEN + kk0 + sub * 8;
      vf.u[0] = *(const uint4*)vptr;
      vf.u[1] = *(const uint4*)(vptr + 16);
      oacc[t] = wmma_bf16(pf.v, vf.v, oacc[t]);
    }
    __syncthreads();
  }

  const int orow0 = b * M + mt + sub * 8;
#pragma unroll
  for (int t = 0; t < 4; ++t) {
    CDfrag c; c.v = oacc[t];
#pragma unroll
    for (int r = 0; r < 8; ++r) {
      const float inv = 1.0f / lrun[r];
      Oout[(size_t)(orow0 + r) * H + hd0 + t * 16 + l16] = (__bf16)(c.f[r] * inv);
    }
  }
}

// ----------------------------------------------------------------- launcher
extern "C" void kernel_launch(void* const* d_in, const int* in_sizes, int n_in,
                              void* d_out, int out_size, void* d_ws, size_t ws_size,
                              hipStream_t stream) {
  const float* query = (const float*)d_in[0];
  const float* key   = (const float*)d_in[1];
  const float* value = (const float*)d_in[2];
  const float* pose  = (const float*)d_in[3];
  const float* ubias = (const float*)d_in[4];
  const float* vbias = (const float*)d_in[5];
  const float* Wq    = (const float*)d_in[6];
  const float* Wk    = (const float*)d_in[7];
  const float* Wv    = (const float*)d_in[8];
  const float* Wr    = (const float*)d_in[9];
  const float* Wo    = (const float*)d_in[10];

  const int H    = in_sizes[4];          // NH * hd = 1024
  const int KLEN = in_sizes[3] / H;      // 2048
  const int BK   = in_sizes[1] / H;      // B*KLEN
  const int B    = BK / KLEN;            // 2
  const int BM   = in_sizes[0] / H;      // B*M
  const int M    = BM / B;               // 1024

  __bf16* ws = (__bf16*)d_ws;
  size_t off = 0;
  auto take = [&](size_t n) { __bf16* p = ws + off; off += (n + 15) & ~(size_t)15; return p; };

  __bf16* qb   = take((size_t)BM * H);
  __bf16* kbuf = take((size_t)BK * H);
  __bf16* vbuf = take((size_t)BK * H);
  __bf16* pb   = take((size_t)KLEN * H);
  __bf16* wqb  = take((size_t)H * H);
  __bf16* wkb  = take((size_t)H * H);
  __bf16* wvb  = take((size_t)H * H);
  __bf16* wrb  = take((size_t)H * H);
  __bf16* wob  = take((size_t)H * H);
  __bf16* Qp   = take((size_t)BM * H);
  __bf16* Kp   = take((size_t)BK * H);
  __bf16* Vt   = take((size_t)BK * H);
  __bf16* Rp   = take((size_t)KLEN * H);
  __bf16* AO   = take((size_t)BM * H);

  auto cast = [&](const float* s, __bf16* d, size_t n) {
    int blocks = (int)((n + 2047) / 2048);
    if (blocks > 4096) blocks = 4096;
    cast_f32_bf16<<<blocks, 256, 0, stream>>>(s, d, n);
  };
  cast(query, qb,   (size_t)BM * H);
  cast(key,   kbuf, (size_t)BK * H);
  cast(value, vbuf, (size_t)BK * H);
  cast(pose,  pb,   (size_t)KLEN * H);
  cast(Wq, wqb, (size_t)H * H);
  cast(Wk, wkb, (size_t)H * H);
  cast(Wv, wvb, (size_t)H * H);
  cast(Wr, wrb, (size_t)H * H);
  cast(Wo, wob, (size_t)H * H);

  auto gemm = [&](const __bf16* A, const __bf16* W, void* C, int m, int n, int k,
                  int mode, int klen) {
    const int waves = (m / 64) * (n / 64);
    gemm_bf16_wmma<<<(waves + 7) / 8, 256, 0, stream>>>(A, W, C, m, n, k, mode, klen);
  };
  gemm(qb,   wqb, Qp, BM,   H, H, 0, 0);     // Q = query @ Wq^T
  gemm(kbuf, wkb, Kp, BK,   H, H, 0, 0);     // K = key   @ Wk^T
  gemm(vbuf, wvb, Vt, BK,   H, H, 1, KLEN);  // V = value @ Wv^T, stored [B][H][KLEN]
  gemm(pb,   wrb, Rp, KLEN, H, H, 0, 0);     // R = pos   @ Wr^T

  relattn_wmma<<<B * NH * (M / 16), 32, 0, stream>>>(Qp, Kp, Vt, Rp, ubias, vbias,
                                                     AO, M, KLEN, H);

  gemm(AO, wob, d_out, BM, H, H, 2, 0);      // out = attn @ Wo^T (fp32 store)
}
